// MessagePassing_81484119539872
// MI455X (gfx1250) — matrix operand order repaired and checked by reading, not compiled, not run on hardware
//
#include <hip/hip_runtime.h>
#include <hip/hip_bf16.h>
#include <math.h>

typedef __attribute__((ext_vector_type(16))) _Float16 v16h;
typedef __attribute__((ext_vector_type(8)))  _Float16 v8h;
typedef __attribute__((ext_vector_type(8)))  float    v8f;

#define C_S_      0.3826834323650898f
#define C_X_      0.9238795325112867f
#define INV_SQRT3 0.5773502691896258f
#define INV_NEI   0.25f
#define RS8       0.35355339059327373f
#define RS16      0.25f
#define RS32      0.17677669529663687f
#define RS40      0.15811388300841897f
#define RS64      0.125f
#define EPSV      1e-5f

// fast activations: v_exp_f32 + v_rcp_f32 (no IEEE div expansion)
__device__ __forceinline__ float sigm_f(float x){
  return __builtin_amdgcn_rcpf(1.f + __expf(-x));
}
__device__ __forceinline__ float silu_f(float x){ return x * sigm_f(x); }

#define WMMA_F16(A,B,C) __builtin_amdgcn_wmma_f32_16x16x32_f16(false,(A),false,(B),(short)0,(C),false,false)

// ===========================================================================
// Edge kernels: fcnet via WMMA chain + message construction + atomic scatter.
// One wave (32 lanes) handles a 16-edge tile; 4 waves / block (128 threads).
//
// LDS layouts (all fragment accesses are contiguous, 16/32B aligned):
//  - sWp : A-fragments pre-packed per (tile, lane): sWp[t*512 + lane*16 + h]
//          halves; one v16h load per fragment (2x ds_load_b128).
//  - sH  : per-wave hidden staging, row-major [edge n][channel k] (16x64 f16);
//          B fragment for (kk0) = v16h at &sH[n*64 + kk0 + g*16];
//          D tile store = v8h at &sH[n*64 + mt*16 + g*8].
//  - sWout: per-wave MLP output, row-major [edge n][channel c] f32; v8f store.
// ===========================================================================
template <int LAYER>
__global__ __launch_bounds__(128) void edge_kernel(
    const float* __restrict__ escal,      // (E,8)
    const int*   __restrict__ esrc,
    const int*   __restrict__ edst,
    const float* __restrict__ eattr,      // (E,4)
    const float* __restrict__ w1,         // (8,64)
    const float* __restrict__ w2,         // (64,64)
    const float* __restrict__ w3,         // (64,K3)
    const float* __restrict__ gnode0,     // L1:(N,16)  L2/L3:(N,32)
    const float* __restrict__ gnode1,     // L1:unused  L2/L3:(N,8,3)
    float* __restrict__ mid0,             // L1:(N,16)  L2/L3:(N,40)
    float* __restrict__ mid1)             // L1:(N,16,3) L2:(N,40,3) L3:unused
{
  constexpr int K3  = (LAYER == 1) ? 32 : ((LAYER == 2) ? 80 : 40);
  constexpr int MT3 = K3 / 16;
  constexpr int NT  = 12 + 2 * MT3;      // # A-fragment tiles

  __shared__ alignas(32) _Float16 sWp[NT * 512];     // packed A fragments
  __shared__ alignas(32) _Float16 sH[4][16 * 64];    // per-wave staging
  __shared__ alignas(32) float    sWout[4][16 * K3]; // per-wave MLP output

  const int tid  = threadIdx.x;
  const int wv   = tid >> 5;
  const int lane = tid & 31;

  // ---- pack A fragments (W^T tiles) into lane-major order, f32 -> f16 ----
  // tile t: [0,4)   -> GEMM1: mt=t,        kk0=0,        W1 (Cin=8, Cout=64)
  //         [4,12)  -> GEMM2: mt=(t-4)/2,  kk0=32*((t-4)&1), W2 (64,64)
  //         [12,NT) -> GEMM3: mt=(t-12)/2, kk0=32*((t-12)&1), W3 (64,K3)
  for (int idx = tid; idx < NT * 512; idx += 128) {
    const int t   = idx >> 9;
    const int rem = idx & 511;
    const int l   = rem >> 4, h = rem & 15;
    const int M = l & 15, g = l >> 4;
    const float* Wsrc; int Cout, Cin, mt, kk0;
    if (t < 4)       { mt = t;            kk0 = 0;                 Wsrc = w1; Cout = 64; Cin = 8;  }
    else if (t < 12) { mt = (t - 4) >> 1; kk0 = 32 * ((t - 4) & 1);  Wsrc = w2; Cout = 64; Cin = 64; }
    else             { mt = (t - 12) >> 1; kk0 = 32 * ((t - 12) & 1); Wsrc = w3; Cout = K3; Cin = 64; }
    const int k = kk0 + g * 16 + h;
    const float v = (k < Cin) ? Wsrc[k * Cout + mt * 16 + M] : 0.f;
    sWp[idx] = (_Float16)v;
  }

  const long base = ((long)blockIdx.x * 4 + wv) * 16;  // E divisible by 64
  const int nn = lane & 15, gg = lane >> 4;
  _Float16* sHw = &sH[wv][0];
  float*    sWo = &sWout[wv][0];

  // ---- stage S^T zero-padded to K=32 into sH (row-major [n][k]) ----
  {
    const v8h z8 = {};
    if (gg == 0) {
      const float4* sp = (const float4*)(escal + (base + nn) * 8);
      const float4 s0 = sp[0], s1 = sp[1];
      v8h hv;
      hv[0] = (_Float16)s0.x; hv[1] = (_Float16)s0.y;
      hv[2] = (_Float16)s0.z; hv[3] = (_Float16)s0.w;
      hv[4] = (_Float16)s1.x; hv[5] = (_Float16)s1.y;
      hv[6] = (_Float16)s1.z; hv[7] = (_Float16)s1.w;
      *(v8h*)&sHw[nn * 64 + 0] = hv;
      *(v8h*)&sHw[nn * 64 + 8] = z8;
    } else {
      *(v8h*)&sHw[nn * 64 + 16] = z8;
      *(v8h*)&sHw[nn * 64 + 24] = z8;
    }
  }
  __syncthreads();

  // ---- GEMM1: H1^T = silu( (W1^T @ S^T) * RS8 ), K padded to 32 ----
  {
    const v16h b0 = *(const v16h*)&sHw[nn * 64 + gg * 16];
#pragma unroll
    for (int mt = 0; mt < 4; ++mt) {
      const v16h a = *(const v16h*)&sWp[mt * 512 + lane * 16];
      v8f acc = {};
      acc = WMMA_F16(a, b0, acc);
      v8h hv;
#pragma unroll
      for (int r = 0; r < 8; ++r) hv[r] = (_Float16)silu_f(acc[r] * RS8);
      *(v8h*)&sHw[nn * 64 + mt * 16 + gg * 8] = hv;
    }
  }
  __syncthreads();

  // ---- GEMM2: H2^T = silu( (W2^T @ H1^T) * RS64 ), K=64 ----
  {
    const v16h hb0 = *(const v16h*)&sHw[nn * 64 + 0  + gg * 16];
    const v16h hb1 = *(const v16h*)&sHw[nn * 64 + 32 + gg * 16];
    __syncthreads();   // H1 fully read before overwrite with H2
#pragma unroll
    for (int mt = 0; mt < 4; ++mt) {
      const v16h a0 = *(const v16h*)&sWp[(4 + mt * 2 + 0) * 512 + lane * 16];
      const v16h a1 = *(const v16h*)&sWp[(4 + mt * 2 + 1) * 512 + lane * 16];
      v8f acc = {};
      acc = WMMA_F16(a0, hb0, acc);
      acc = WMMA_F16(a1, hb1, acc);
      v8h hv;
#pragma unroll
      for (int r = 0; r < 8; ++r) hv[r] = (_Float16)silu_f(acc[r] * RS64);
      *(v8h*)&sHw[nn * 64 + mt * 16 + gg * 8] = hv;
    }
  }
  __syncthreads();

  // ---- GEMM3: w^T = (W3^T @ H2^T) * RS64 -> f32 row-major [n][K3] ----
  {
    const v16h cb0 = *(const v16h*)&sHw[nn * 64 + 0  + gg * 16];
    const v16h cb1 = *(const v16h*)&sHw[nn * 64 + 32 + gg * 16];
#pragma unroll
    for (int mt = 0; mt < MT3; ++mt) {
      const v16h a0 = *(const v16h*)&sWp[(12 + mt * 2 + 0) * 512 + lane * 16];
      const v16h a1 = *(const v16h*)&sWp[(12 + mt * 2 + 1) * 512 + lane * 16];
      v8f acc = {};
      acc = WMMA_F16(a0, cb0, acc);
      acc = WMMA_F16(a1, cb1, acc);
      v8f av;
#pragma unroll
      for (int r = 0; r < 8; ++r) av[r] = acc[r] * RS64;
      *(v8f*)&sWo[nn * K3 + mt * 16 + gg * 8] = av;
    }
  }
  __syncthreads();

  // ---- message phase: lane pair (nn, nn+16) shares edge, splits channels --
  const long e = base + nn;
  const int src = esrc[e], dst = edst[e];
  const float e0v = eattr[e * 4 + 0];
  const float ex  = eattr[e * 4 + 1];
  const float ey  = eattr[e * 4 + 2];
  const float ez  = eattr[e * 4 + 3];
  const float* wo = &sWo[nn * K3];   // this edge's MLP output, contiguous

  if constexpr (LAYER == 1) {
#pragma unroll
    for (int r = 0; r < 8; ++r) {
      const int c = gg * 8 + r;
      const float g  = gnode0[(long)src * 16 + c];
      const float w0 = wo[c];        // w[:, :16]
      const float wm = wo[16 + c];   // w[:, 16:32]
      atomicAdd(&mid0[(long)dst * 16 + c], w0 * g * e0v);
      const float t = wm * g;
      float* p = &mid1[((long)dst * 16 + c) * 3];
      atomicAdd(p + 0, t * ex); atomicAdd(p + 1, t * ey); atomicAdd(p + 2, t * ez);
    }
  } else if constexpr (LAYER == 2) {
#pragma unroll
    for (int r = 0; r < 16; ++r) {
      const int c = gg * 16 + r;
      const float g  = gnode0[(long)src * 32 + c];
      const float wa = wo[c];        // w[:, :32]   -> m0a
      const float wm = wo[40 + c];   // w[:, 40:72] -> m1a
      atomicAdd(&mid0[(long)dst * 40 + c], wa * g * e0v);
      const float t = wm * g;
      float* p = &mid1[((long)dst * 40 + c) * 3];
      atomicAdd(p + 0, t * ex); atomicAdd(p + 1, t * ey); atomicAdd(p + 2, t * ez);
    }
#pragma unroll
    for (int r = 0; r < 4; ++r) {
      const int u = gg * 4 + r;
      const float* g1 = &gnode1[((long)src * 8 + u) * 3];
      const float gx = g1[0], gy = g1[1], gz = g1[2];
      const float wb = wo[32 + u];   // w[:, 32:40] -> m0b
      const float wc = wo[72 + u];   // w[:, 72:80] -> m1b
      const float dot = gx * ex + gy * ey + gz * ez;
      atomicAdd(&mid0[(long)dst * 40 + 32 + u], wb * dot * INV_SQRT3);
      const float t = wc * e0v;
      float* p = &mid1[((long)dst * 40 + 32 + u) * 3];
      atomicAdd(p + 0, t * gx); atomicAdd(p + 1, t * gy); atomicAdd(p + 2, t * gz);
    }
  } else {  // LAYER == 3
#pragma unroll
    for (int r = 0; r < 16; ++r) {
      const int c = gg * 16 + r;
      const float g  = gnode0[(long)src * 32 + c];
      atomicAdd(&mid0[(long)dst * 40 + c], wo[c] * g * e0v);
    }
#pragma unroll
    for (int r = 0; r < 4; ++r) {
      const int u = gg * 4 + r;
      const float* g1 = &gnode1[((long)src * 8 + u) * 3];
      const float dot = g1[0] * ex + g1[1] * ey + g1[2] * ez;
      atomicAdd(&mid0[(long)dst * 40 + 32 + u], wo[32 + u] * dot * INV_SQRT3);
    }
  }
}

// ===========================================================================
// Node-side kernels (small GEMMs, batchnorm stats, gating) — VALU.
// ===========================================================================
__global__ __launch_bounds__(256) void node_pre1_kernel(
    const float* __restrict__ nf, const float* __restrict__ na,
    const float* __restrict__ Wsc1, const float* __restrict__ Wl11,
    float* __restrict__ sc0_1, float* __restrict__ h1node,
    float* __restrict__ mid0_1, float* __restrict__ mid1_1,
    float* __restrict__ stats1, int N)
{
  const long n = (long)blockIdx.x * blockDim.x + threadIdx.x;
  if (n < 128) stats1[n] = 0.f;
  if (n >= N) return;
  const float a = na[n];
  float x[16];
#pragma unroll
  for (int i = 0; i < 16; ++i) x[i] = nf[n * 16 + i];
  for (int j = 0; j < 40; ++j) {
    float s = 0.f;
#pragma unroll
    for (int i = 0; i < 16; ++i) s += x[i] * Wsc1[i * 40 + j];
    sc0_1[n * 40 + j] = a * s * RS16;
  }
  for (int j = 0; j < 16; ++j) {
    float s = 0.f;
#pragma unroll
    for (int i = 0; i < 16; ++i) s += x[i] * Wl11[i * 16 + j];
    h1node[n * 16 + j] = a * s * RS16;
  }
#pragma unroll
  for (int c = 0; c < 16; ++c) mid0_1[n * 16 + c] = 0.f;
  for (int k = 0; k < 48; ++k) mid1_1[n * 48 + k] = 0.f;
}

__global__ __launch_bounds__(256) void node_bn1_kernel(
    const float* __restrict__ mid0, const float* __restrict__ mid1,
    const float* __restrict__ na, const float* __restrict__ sc0_1,
    const float* __restrict__ Wl210, const float* __restrict__ Wl211,
    float* __restrict__ z0buf, float* __restrict__ z1buf,
    float* __restrict__ stats, int N)
{
  __shared__ float sacc[88];
  const int tid = threadIdx.x;
  if (tid < 88) sacc[tid] = 0.f;
  __syncthreads();
  const long n = (long)blockIdx.x * blockDim.x + tid;
  if (n < N) {
    const float a = na[n];
    float m0[16];
#pragma unroll
    for (int i = 0; i < 16; ++i) m0[i] = mid0[n * 16 + i] * INV_NEI;
    for (int j = 0; j < 40; ++j) {
      float s = 0.f;
#pragma unroll
      for (int i = 0; i < 16; ++i) s += m0[i] * Wl210[i * 40 + j];
      const float z = C_S_ * sc0_1[n * 40 + j] + C_X_ * (a * s * RS16);
      z0buf[n * 40 + j] = z;
      atomicAdd(&sacc[j], z);
      atomicAdd(&sacc[40 + j], z * z);
    }
    float z1a[8][3];
#pragma unroll
    for (int v = 0; v < 8; ++v) { z1a[v][0] = 0.f; z1a[v][1] = 0.f; z1a[v][2] = 0.f; }
    for (int u = 0; u < 16; ++u) {
      const float mx = mid1[(n * 16 + u) * 3 + 0] * INV_NEI;
      const float my = mid1[(n * 16 + u) * 3 + 1] * INV_NEI;
      const float mz = mid1[(n * 16 + u) * 3 + 2] * INV_NEI;
#pragma unroll
      for (int v = 0; v < 8; ++v) {
        const float w = Wl211[u * 8 + v];
        z1a[v][0] += mx * w; z1a[v][1] += my * w; z1a[v][2] += mz * w;
      }
    }
#pragma unroll
    for (int v = 0; v < 8; ++v) {
      const float zx = a * z1a[v][0] * RS16;
      const float zy = a * z1a[v][1] * RS16;
      const float zz = a * z1a[v][2] * RS16;
      z1buf[(n * 8 + v) * 3 + 0] = zx;
      z1buf[(n * 8 + v) * 3 + 1] = zy;
      z1buf[(n * 8 + v) * 3 + 2] = zz;
      atomicAdd(&sacc[80 + v], (zx * zx + zy * zy + zz * zz) * (1.f / 3.f));
    }
  }
  __syncthreads();
  if (tid < 88) atomicAdd(&stats[tid], sacc[tid]);
}

__global__ __launch_bounds__(256) void node_gate1_kernel(
    const float* __restrict__ z0buf, const float* __restrict__ z1buf,
    const float* __restrict__ stats1, const float* __restrict__ na,
    const float* __restrict__ Wsc20, const float* __restrict__ Wsc21,
    const float* __restrict__ Wl120, const float* __restrict__ Wl121,
    float* __restrict__ sc0_2, float* __restrict__ sc1v,
    float* __restrict__ h0buf, float* __restrict__ h1buf,
    float* __restrict__ mid0_2, float* __restrict__ mid1_2,
    float* __restrict__ stats2, int N)
{
  const long n = (long)blockIdx.x * blockDim.x + threadIdx.x;
  if (n < 128) stats2[n] = 0.f;
  if (n >= N) return;
  const float a = na[n];
  const float invN = 1.f / (float)N;
  float x0[32], gt[8];
  for (int c = 0; c < 40; ++c) {
    const float mu  = stats1[c] * invN;
    const float var = stats1[40 + c] * invN - mu * mu;
    const float z   = (z0buf[n * 40 + c] - mu) * rsqrtf(var + EPSV);
    if (c < 32) x0[c] = silu_f(z); else gt[c - 32] = sigm_f(z);
  }
  float x1[8][3];
#pragma unroll
  for (int u = 0; u < 8; ++u) {
    const float f = rsqrtf(stats1[80 + u] * invN + EPSV) * gt[u];
#pragma unroll
    for (int c = 0; c < 3; ++c) x1[u][c] = z1buf[(n * 8 + u) * 3 + c] * f;
  }
  for (int j = 0; j < 40; ++j) {
    float s = 0.f;
#pragma unroll
    for (int i = 0; i < 32; ++i) s += x0[i] * Wsc20[i * 40 + j];
    sc0_2[n * 40 + j] = a * s * RS32;
  }
  for (int j = 0; j < 32; ++j) {
    float s = 0.f;
#pragma unroll
    for (int i = 0; i < 32; ++i) s += x0[i] * Wl120[i * 32 + j];
    h0buf[n * 32 + j] = a * s * RS32;
  }
#pragma unroll
  for (int v = 0; v < 8; ++v) {
    float sx = 0.f, sy = 0.f, sz = 0.f, tx = 0.f, ty = 0.f, tz = 0.f;
#pragma unroll
    for (int u = 0; u < 8; ++u) {
      const float ws = Wsc21[u * 8 + v], wl = Wl121[u * 8 + v];
      sx += x1[u][0] * ws; sy += x1[u][1] * ws; sz += x1[u][2] * ws;
      tx += x1[u][0] * wl; ty += x1[u][1] * wl; tz += x1[u][2] * wl;
    }
    sc1v[(n * 8 + v) * 3 + 0] = a * sx * RS8;
    sc1v[(n * 8 + v) * 3 + 1] = a * sy * RS8;
    sc1v[(n * 8 + v) * 3 + 2] = a * sz * RS8;
    h1buf[(n * 8 + v) * 3 + 0] = a * tx * RS8;
    h1buf[(n * 8 + v) * 3 + 1] = a * ty * RS8;
    h1buf[(n * 8 + v) * 3 + 2] = a * tz * RS8;
  }
#pragma unroll
  for (int c = 0; c < 40; ++c) mid0_2[n * 40 + c] = 0.f;
  for (int k = 0; k < 120; ++k) mid1_2[n * 120 + k] = 0.f;
}

__global__ __launch_bounds__(256) void node_bn2_kernel(
    const float* __restrict__ mid0, const float* __restrict__ mid1,
    const float* __restrict__ na,
    const float* __restrict__ sc0_2, const float* __restrict__ sc1v,
    const float* __restrict__ Wl220, const float* __restrict__ Wl221,
    float* __restrict__ z0buf, float* __restrict__ z1buf,
    float* __restrict__ stats, int N)
{
  __shared__ float sacc[88];
  const int tid = threadIdx.x;
  if (tid < 88) sacc[tid] = 0.f;
  __syncthreads();
  const long n = (long)blockIdx.x * blockDim.x + tid;
  if (n < N) {
    const float a = na[n];
    float z0a[40];
#pragma unroll
    for (int j = 0; j < 40; ++j) z0a[j] = 0.f;
    for (int i = 0; i < 40; ++i) {
      const float m = mid0[n * 40 + i] * INV_NEI;
#pragma unroll
      for (int j = 0; j < 40; ++j) z0a[j] += m * Wl220[i * 40 + j];
    }
    for (int j = 0; j < 40; ++j) {
      const float z = C_S_ * sc0_2[n * 40 + j] + C_X_ * (a * z0a[j] * RS40);
      z0buf[n * 40 + j] = z;
      atomicAdd(&sacc[j], z);
      atomicAdd(&sacc[40 + j], z * z);
    }
    float z1a[8][3];
#pragma unroll
    for (int v = 0; v < 8; ++v) { z1a[v][0] = 0.f; z1a[v][1] = 0.f; z1a[v][2] = 0.f; }
    for (int u = 0; u < 40; ++u) {
      const float mx = mid1[(n * 40 + u) * 3 + 0] * INV_NEI;
      const float my = mid1[(n * 40 + u) * 3 + 1] * INV_NEI;
      const float mz = mid1[(n * 40 + u) * 3 + 2] * INV_NEI;
#pragma unroll
      for (int v = 0; v < 8; ++v) {
        const float w = Wl221[u * 8 + v];
        z1a[v][0] += mx * w; z1a[v][1] += my * w; z1a[v][2] += mz * w;
      }
    }
#pragma unroll
    for (int v = 0; v < 8; ++v) {
      const float zx = C_S_ * sc1v[(n * 8 + v) * 3 + 0] + C_X_ * (a * z1a[v][0] * RS40);
      const float zy = C_S_ * sc1v[(n * 8 + v) * 3 + 1] + C_X_ * (a * z1a[v][1] * RS40);
      const float zz = C_S_ * sc1v[(n * 8 + v) * 3 + 2] + C_X_ * (a * z1a[v][2] * RS40);
      z1buf[(n * 8 + v) * 3 + 0] = zx;
      z1buf[(n * 8 + v) * 3 + 1] = zy;
      z1buf[(n * 8 + v) * 3 + 2] = zz;
      atomicAdd(&sacc[80 + v], (zx * zx + zy * zy + zz * zz) * (1.f / 3.f));
    }
  }
  __syncthreads();
  if (tid < 88) atomicAdd(&stats[tid], sacc[tid]);
}

__global__ __launch_bounds__(256) void node_gate2_kernel(
    const float* __restrict__ z0buf, const float* __restrict__ z1buf,
    const float* __restrict__ stats2, const float* __restrict__ na,
    const float* __restrict__ Wsc30, const float* __restrict__ Wl130,
    const float* __restrict__ Wl131,
    float* __restrict__ sc0_3, float* __restrict__ h0buf,
    float* __restrict__ h1buf, float* __restrict__ mid0_3, int N)
{
  const long n = (long)blockIdx.x * blockDim.x + threadIdx.x;
  if (n >= N) return;
  const float a = na[n];
  const float invN = 1.f / (float)N;
  float x0[32], gt[8];
  for (int c = 0; c < 40; ++c) {
    const float mu  = stats2[c] * invN;
    const float var = stats2[40 + c] * invN - mu * mu;
    const float z   = (z0buf[n * 40 + c] - mu) * rsqrtf(var + EPSV);
    if (c < 32) x0[c] = silu_f(z); else gt[c - 32] = sigm_f(z);
  }
  float x1[8][3];
#pragma unroll
  for (int u = 0; u < 8; ++u) {
    const float f = rsqrtf(stats2[80 + u] * invN + EPSV) * gt[u];
#pragma unroll
    for (int c = 0; c < 3; ++c) x1[u][c] = z1buf[(n * 8 + u) * 3 + c] * f;
  }
  for (int j = 0; j < 8; ++j) {
    float s = 0.f;
#pragma unroll
    for (int i = 0; i < 32; ++i) s += x0[i] * Wsc30[i * 8 + j];
    sc0_3[n * 8 + j] = a * s * RS32;
  }
  for (int j = 0; j < 32; ++j) {
    float s = 0.f;
#pragma unroll
    for (int i = 0; i < 32; ++i) s += x0[i] * Wl130[i * 32 + j];
    h0buf[n * 32 + j] = a * s * RS32;
  }
#pragma unroll
  for (int v = 0; v < 8; ++v) {
    float tx = 0.f, ty = 0.f, tz = 0.f;
#pragma unroll
    for (int u = 0; u < 8; ++u) {
      const float wl = Wl131[u * 8 + v];
      tx += x1[u][0] * wl; ty += x1[u][1] * wl; tz += x1[u][2] * wl;
    }
    h1buf[(n * 8 + v) * 3 + 0] = a * tx * RS8;
    h1buf[(n * 8 + v) * 3 + 1] = a * ty * RS8;
    h1buf[(n * 8 + v) * 3 + 2] = a * tz * RS8;
  }
#pragma unroll
  for (int c = 0; c < 40; ++c) mid0_3[n * 40 + c] = 0.f;
}

__global__ __launch_bounds__(256) void node_out_kernel(
    const float* __restrict__ mid0, const float* __restrict__ sc0_3,
    const float* __restrict__ na, const float* __restrict__ Wl230,
    float* __restrict__ out, int N)
{
  const long n = (long)blockIdx.x * blockDim.x + threadIdx.x;
  if (n >= N) return;
  const float a = na[n];
  float m[40];
#pragma unroll
  for (int c = 0; c < 40; ++c) m[c] = mid0[n * 40 + c] * INV_NEI;
#pragma unroll
  for (int v = 0; v < 8; ++v) {
    float s = 0.f;
#pragma unroll
    for (int c = 0; c < 40; ++c) s += m[c] * Wl230[c * 8 + v];
    out[n * 8 + v] = C_S_ * sc0_3[n * 8 + v] + C_X_ * (a * s * RS40);
  }
}

// ===========================================================================
extern "C" void kernel_launch(void* const* d_in, const int* in_sizes, int n_in,
                              void* d_out, int out_size, void* d_ws, size_t ws_size,
                              hipStream_t stream) {
  (void)n_in; (void)out_size; (void)ws_size;
  const float* nf    = (const float*)d_in[0];
  const float* na    = (const float*)d_in[1];
  const int*   esrc  = (const int*)  d_in[2];
  const int*   edst  = (const int*)  d_in[3];
  const float* eattr = (const float*)d_in[4];
  const float* escal = (const float*)d_in[5];
  // params in PARAM_SHAPES order, starting at d_in[6]
  const float* Wsc1   = (const float*)d_in[6];
  const float* Wl11   = (const float*)d_in[7];
  const float* fc1w1  = (const float*)d_in[8];
  const float* fc1w2  = (const float*)d_in[9];
  const float* fc1w3  = (const float*)d_in[10];
  const float* Wl210  = (const float*)d_in[11];
  const float* Wl211  = (const float*)d_in[12];
  const float* Wsc20  = (const float*)d_in[13];
  const float* Wsc21  = (const float*)d_in[14];
  const float* Wl120  = (const float*)d_in[15];
  const float* Wl121  = (const float*)d_in[16];
  const float* fc2w1  = (const float*)d_in[17];
  const float* fc2w2  = (const float*)d_in[18];
  const float* fc2w3  = (const float*)d_in[19];
  const float* Wl220  = (const float*)d_in[20];
  const float* Wl221  = (const float*)d_in[21];
  const float* Wsc30  = (const float*)d_in[22];
  const float* Wl130  = (const float*)d_in[23];
  const float* Wl131  = (const float*)d_in[24];
  const float* fc3w1  = (const float*)d_in[25];
  const float* fc3w2  = (const float*)d_in[26];
  const float* fc3w3  = (const float*)d_in[27];
  const float* Wl230  = (const float*)d_in[28];

  const int N = in_sizes[0] / 16;
  const int E = in_sizes[2];

  // workspace layout (floats)
  float* ws = (float*)d_ws;
  size_t o = 0;
  float* h1node = ws + o; o += (size_t)N * 16;
  float* sc0_1  = ws + o; o += (size_t)N * 40;
  float* mid0_1 = ws + o; o += (size_t)N * 16;
  float* mid1_1 = ws + o; o += (size_t)N * 48;
  float* z0buf  = ws + o; o += (size_t)N * 40;
  float* z1buf  = ws + o; o += (size_t)N * 24;
  float* stats1 = ws + o; o += 128;
  float* stats2 = ws + o; o += 128;
  float* sc0_2  = ws + o; o += (size_t)N * 40;
  float* sc1v   = ws + o; o += (size_t)N * 24;
  float* h0buf  = ws + o; o += (size_t)N * 32;
  float* h1buf  = ws + o; o += (size_t)N * 24;
  float* mid0_2 = ws + o; o += (size_t)N * 40;   // reused as mid0_3
  float* mid1_2 = ws + o; o += (size_t)N * 120;
  float* sc0_3  = ws + o; o += (size_t)N * 8;

  const int nodeBlocks = (N + 255) / 256;
  const int edgeBlocks = E / 64;   // 4 waves x 16 edges per block (E % 64 == 0)

  // ---- layer 1 ----
  node_pre1_kernel<<<nodeBlocks, 256, 0, stream>>>(
      nf, na, Wsc1, Wl11, sc0_1, h1node, mid0_1, mid1_1, stats1, N);
  edge_kernel<1><<<edgeBlocks, 128, 0, stream>>>(
      escal, esrc, edst, eattr, fc1w1, fc1w2, fc1w3,
      h1node, nullptr, mid0_1, mid1_1);
  node_bn1_kernel<<<nodeBlocks, 256, 0, stream>>>(
      mid0_1, mid1_1, na, sc0_1, Wl210, Wl211, z0buf, z1buf, stats1, N);
  node_gate1_kernel<<<nodeBlocks, 256, 0, stream>>>(
      z0buf, z1buf, stats1, na, Wsc20, Wsc21, Wl120, Wl121,
      sc0_2, sc1v, h0buf, h1buf, mid0_2, mid1_2, stats2, N);

  // ---- layer 2 ----
  edge_kernel<2><<<edgeBlocks, 128, 0, stream>>>(
      escal, esrc, edst, eattr, fc2w1, fc2w2, fc2w3,
      h0buf, h1buf, mid0_2, mid1_2);
  node_bn2_kernel<<<nodeBlocks, 256, 0, stream>>>(
      mid0_2, mid1_2, na, sc0_2, sc1v, Wl220, Wl221, z0buf, z1buf, stats2, N);
  node_gate2_kernel<<<nodeBlocks, 256, 0, stream>>>(
      z0buf, z1buf, stats2, na, Wsc30, Wl130, Wl131,
      sc0_3, h0buf, h1buf, mid0_2 /* = mid0_3 */, N);

  // ---- layer 3 ----
  edge_kernel<3><<<edgeBlocks, 128, 0, stream>>>(
      escal, esrc, edst, eattr, fc3w1, fc3w2, fc3w3,
      h0buf, h1buf, mid0_2, nullptr);
  node_out_kernel<<<nodeBlocks, 256, 0, stream>>>(
      mid0_2, sc0_3, na, Wl230, (float*)d_out, N);
}
